// MultiHeadAttention_35613868819021
// MI455X (gfx1250) — matrix operand order repaired and checked by reading, not compiled, not run on hardware
//
#include <hip/hip_runtime.h>

#define SEQ     2048
#define HID     2048
#define NH      16
#define HD      128
#define THREE_H 6144
#define LDT     36   // LDS tile leading dim (halves): 32 + 4 pad (TDM pad 2 DW / 16 DW)

typedef __attribute__((ext_vector_type(16))) __bf16 v16bf;
typedef __attribute__((ext_vector_type(8)))  float  v8f;
typedef __attribute__((ext_vector_type(4)))  unsigned int v4u;
typedef __attribute__((ext_vector_type(8)))  int  v8i;
typedef __attribute__((ext_vector_type(4)))  int  v4i;

struct FragU { unsigned int u[8]; };

__device__ __forceinline__ unsigned short f2bf(float f) {
  unsigned int u = __float_as_uint(f);
  u += 0x7FFFu + ((u >> 16) & 1u);          // round-to-nearest-even
  return (unsigned short)(u >> 16);
}

__device__ __forceinline__ v8f zero8() {
  v8f z = {0.f, 0.f, 0.f, 0.f, 0.f, 0.f, 0.f, 0.f};
  return z;
}

__device__ __forceinline__ v8f wmma_bf16(v16bf a, v16bf b, v8f c) {
  return __builtin_amdgcn_wmma_f32_16x16x32_bf16(false, a, false, b, (short)0, c, false, false);
}

// A-fragment 16x32: lane(0-15)=M row, lane half selects K sub-range
// halves 0..15 -> K = kb..kb+15 ; kb = (lane&16)?16:0  (contiguous)
__device__ __forceinline__ v16bf load_fragA(const unsigned short* base, int row0, int ld, int k0) {
  const int lane = threadIdx.x & 31;
  const int m  = lane & 15;
  const int kb = (lane & 16) ? 16 : 0;
  const unsigned int* p =
      reinterpret_cast<const unsigned int*>(base + (size_t)(row0 + m) * ld + k0 + kb);
  FragU f;
#pragma unroll
  for (int j = 0; j < 4; ++j) { f.u[j] = p[j]; f.u[4 + j] = p[4 + j]; }
  return __builtin_bit_cast(v16bf, f);
}

// B-fragment 32x16: element B[k][n] read from row-major W[n][k]
// lane: n = lane&15 ; halves 0..15 -> K = kb..kb+15 ; kb = (lane&16)?16:0  (contiguous)
__device__ __forceinline__ v16bf load_fragB(const unsigned short* base, int n0, int ld, int k0) {
  const int lane = threadIdx.x & 31;
  const int n  = lane & 15;
  const int kb = (lane & 16) ? 16 : 0;
  const unsigned int* p =
      reinterpret_cast<const unsigned int*>(base + (size_t)(n0 + n) * ld + k0 + kb);
  FragU f;
#pragma unroll
  for (int j = 0; j < 8; ++j) f.u[j] = p[j];
  return __builtin_bit_cast(v16bf, f);
}

// ---------------------------------------------------------------------------
// TDM: 2D tile (tile_d1 rows x tile_d0 halves) from row-major bf16 tensor
// (row stride = stride_d0 halves) into LDS at lds_byte_off, with hardware
// padding of +2 DWORDs every 16 DWORDs -> LDS leading dim = 36 halves (LDT).
// D# layout per CDNA5 ISA ch.8 (group0: count/lds/global/type, group1: dims).
// This toolchain exposes the 6-arg builtin; trailing group operands zeroed
// (2D tensor: groups 2/3 disabled).
// ---------------------------------------------------------------------------
__device__ __forceinline__ void tdm_load_2d(unsigned lds_byte_off, const void* gptr,
                                            unsigned tensor_d0, unsigned tensor_d1,
                                            unsigned stride_d0,
                                            unsigned tile_d0, unsigned tile_d1) {
  const unsigned long long ga = (unsigned long long)(size_t)gptr;
  v4u g0;
  g0.x = 1u;                                  // count=1, user mode, no gather
  g0.y = lds_byte_off;                        // lds_addr
  g0.z = (unsigned)ga;                        // global_addr[31:0]
  g0.w = ((unsigned)(ga >> 32) & 0x01FFFFFFu) // global_addr[56:32]
         | (2u << 30);                        // type = 2 ("image")
  v8i g1;
  g1[0] = (int)((1u << 16)      // data_size = 2 bytes
              | (1u << 20)      // pad_enable
              | (3u << 22)      // pad_interval: 16 DWORDs (64B = 32 halves)
              | (1u << 25));    // pad_amount:  2 DWORDs  (+4 halves)
  g1[1] = (int)((tensor_d0 & 0xFFFFu) << 16);                       // tensor_dim0[15:0]
  g1[2] = (int)((tensor_d0 >> 16) | ((tensor_d1 & 0xFFFFu) << 16)); // d0[31:16] | d1[15:0]
  g1[3] = (int)((tensor_d1 >> 16) | (tile_d0 << 16));               // d1[31:16] | tile_dim0
  g1[4] = (int)tile_d1;                                             // tile_dim1 | tile_dim2=0
  g1[5] = (int)stride_d0;                                           // tensor_dim0_stride[31:0]
  g1[6] = 0;                                                        // stride hi | dim1_stride lo
  g1[7] = 0;
  const v4i z4 = {0, 0, 0, 0};                 // groups 2/3 unused (2D tensor)
  const v8i z8 = {0, 0, 0, 0, 0, 0, 0, 0};
  __builtin_amdgcn_tensor_load_to_lds(g0, g1, z4, z4, z8, 0);
}

// ---------------------------------------------------------------------------
// fp32 -> bf16, 4 elements / thread
// ---------------------------------------------------------------------------
__global__ __launch_bounds__(256) void f32_to_bf16_vec4(
    const float* __restrict__ src, unsigned short* __restrict__ dst, int n4) {
  int i = blockIdx.x * 256 + threadIdx.x;
  if (i >= n4) return;
  const float4 v = reinterpret_cast<const float4*>(src)[i];
  uint2 o;
  o.x = (unsigned)f2bf(v.x) | ((unsigned)f2bf(v.y) << 16);
  o.y = (unsigned)f2bf(v.z) | ((unsigned)f2bf(v.w) << 16);
  reinterpret_cast<uint2*>(dst)[i] = o;
}

// ---------------------------------------------------------------------------
// C[M,N] (f32) = A[M,K] (bf16, row-major) * B[N,K]^T (bf16, row-major)
// 128x128 block tile; K chunks of 32 DMA'd into double-buffered LDS by the
// Tensor Data Mover (issued by wave 0, overlapped with WMMA compute).
// ---------------------------------------------------------------------------
__global__ __launch_bounds__(256) void gemm_bf16_nt(
    const unsigned short* __restrict__ A,
    const unsigned short* __restrict__ B,
    float* __restrict__ C,
    int M, int N, int K) {
  __shared__ unsigned short aT[2][128 * LDT];
  __shared__ unsigned short bT[2][128 * LDT];

  const int tid  = threadIdx.x;
  const int lane = tid & 31;
  const int wave = tid >> 5;
  const int wm   = wave & 3;   // 4 row strips of 32
  const int wn   = wave >> 2;  // 2 col strips of 64
  const int rowBlk = blockIdx.y * 128;
  const int colBlk = blockIdx.x * 128;

  v8f acc[2][4];
#pragma unroll
  for (int i = 0; i < 2; ++i)
#pragma unroll
    for (int j = 0; j < 4; ++j) acc[i][j] = zero8();

  const int nK = K >> 5;
  auto issue = [&](int kt, int buf) {
    const int k0 = kt * 32;
    tdm_load_2d((unsigned)(size_t)&aT[buf][0], A + (size_t)rowBlk * K + k0,
                (unsigned)K, (unsigned)M, (unsigned)K, 32u, 128u);
    tdm_load_2d((unsigned)(size_t)&bT[buf][0], B + (size_t)colBlk * K + k0,
                (unsigned)K, (unsigned)N, (unsigned)K, 32u, 128u);
  };

  if (wave == 0) issue(0, 0);

  for (int kt = 0; kt < nK; ++kt) {
    const int buf = kt & 1;
    if (wave == 0) __builtin_amdgcn_s_wait_tensorcnt(0);  // tile[buf] landed
    __syncthreads();
    if (wave == 0 && kt + 1 < nK) issue(kt + 1, buf ^ 1); // prefetch next tile via TDM

    v16bf af[2], bfr[4];
#pragma unroll
    for (int mi = 0; mi < 2; ++mi) af[mi] = load_fragA(&aT[buf][0], wm * 32 + mi * 16, LDT, 0);
#pragma unroll
    for (int ni = 0; ni < 4; ++ni) bfr[ni] = load_fragB(&bT[buf][0], wn * 64 + ni * 16, LDT, 0);
#pragma unroll
    for (int mi = 0; mi < 2; ++mi)
#pragma unroll
      for (int ni = 0; ni < 4; ++ni) acc[mi][ni] = wmma_bf16(af[mi], bfr[ni], acc[mi][ni]);
    __syncthreads();  // all reads of tile[buf] done before TDM may overwrite it
  }

  const int halfOff = (lane & 16) ? 8 : 0;
  const int cLane   = lane & 15;
#pragma unroll
  for (int mi = 0; mi < 2; ++mi)
#pragma unroll
    for (int ni = 0; ni < 4; ++ni)
#pragma unroll
      for (int r = 0; r < 8; ++r) {
        const int row = rowBlk + wm * 32 + mi * 16 + r + halfOff;
        const int col = colBlk + wn * 64 + ni * 16 + cLane;
        C[(size_t)row * N + col] = acc[mi][ni][r];
      }
}

// ---------------------------------------------------------------------------
// Fused per-head RMSNorm + RoPE. Emits bf16 Q,K as [NH][SEQ][HD] and V^T as
// [NH][HD][SEQ] (so PV B-fragments are contiguous).
// ---------------------------------------------------------------------------
__global__ __launch_bounds__(128) void rmsnorm_rope(
    const float* __restrict__ qkv,        // [SEQ][3*HID]
    unsigned short* __restrict__ qh,
    unsigned short* __restrict__ kh,
    unsigned short* __restrict__ vT) {
  const int l = blockIdx.x;
  const int h = blockIdx.y;
  const int d = threadIdx.x;  // 0..127

  const float* row = qkv + (size_t)l * THREE_H + h * HD;
  const float qv = row[d];
  const float kv = row[HID + d];
  const float vv = row[2 * HID + d];

  float sq = qv * qv, sk = kv * kv;
#pragma unroll
  for (int off = 1; off <= 16; off <<= 1) {
    sq += __shfl_xor(sq, off, 32);
    sk += __shfl_xor(sk, off, 32);
  }
  __shared__ float sb[8];
  const int w = threadIdx.x >> 5;
  if ((threadIdx.x & 31) == 0) { sb[w] = sq; sb[4 + w] = sk; }
  __syncthreads();
  const float tq = sb[0] + sb[1] + sb[2] + sb[3];
  const float tk = sb[4] + sb[5] + sb[6] + sb[7];
  const float rq = rsqrtf(tq * (1.0f / HD) + 1e-5f);
  const float rk = rsqrtf(tk * (1.0f / HD) + 1e-5f);

  const float qn = qv * rq;
  const float kn = kv * rk;
  const float qp = __shfl_xor(qn, 1, 32);  // partner in the rotate pair
  const float kp = __shfl_xor(kn, 1, 32);

  // inv_freq = ROPE_BASE^(-(2i)/HD), 2i = d & ~1 ; ln(10000)=9.21034...
  const float freq = __expf(-(float)(d & ~1) * (9.210340371976184f / 128.0f));
  float sn, cs;
  __sincosf((float)l * freq, &sn, &cs);
  // y[2i]   = x[2i+1]*cos - x[2i]*sin ; y[2i+1] = x[2i+1]*sin + x[2i]*cos
  const float sgn = (d & 1) ? 1.f : -1.f;
  const float yq = qp * cs + sgn * qn * sn;
  const float yk = kp * cs + sgn * kn * sn;

  const size_t base = ((size_t)h * SEQ + l) * HD + d;
  qh[base] = f2bf(yq);
  kh[base] = f2bf(yk);
  vT[((size_t)h * HD + d) * SEQ + l] = f2bf(vv);
}

// ---------------------------------------------------------------------------
// Flash attention (causal). Block = 4 waves; wave w owns 16 query rows of a
// 64-row query block. Online softmax; P goes C-layout -> LDS bf16 -> A-frag.
// ---------------------------------------------------------------------------
__global__ __launch_bounds__(128) void flash_attn(
    const unsigned short* __restrict__ qh,
    const unsigned short* __restrict__ kh,
    const unsigned short* __restrict__ vT,
    unsigned short* __restrict__ outb) {     // [SEQ][HID] bf16
  __shared__ unsigned short pLds[4 * 16 * 64];

  const int qb   = blockIdx.x;          // 0..31 (query block of 64)
  const int h    = blockIdx.y;          // head
  const int wave = threadIdx.x >> 5;
  const int lane = threadIdx.x & 31;
  const int halfOff = (lane & 16) ? 8 : 0;
  const int col  = lane & 15;
  const int qRow0 = qb * 64 + wave * 16;

  const unsigned short* Q  = qh + (size_t)h * SEQ * HD;
  const unsigned short* Kp = kh + (size_t)h * SEQ * HD;
  const unsigned short* V  = vT + (size_t)h * HD * SEQ;
  unsigned short* pw = pLds + wave * 16 * 64;

  v16bf qf[4];
#pragma unroll
  for (int c = 0; c < 4; ++c) qf[c] = load_fragA(Q, qRow0, HD, c * 32);

  v8f acc[8];
#pragma unroll
  for (int a = 0; a < 8; ++a) acc[a] = zero8();

  float mrow[8], lrow[8];
#pragma unroll
  for (int r = 0; r < 8; ++r) { mrow[r] = -__builtin_inff(); lrow[r] = 0.f; }

  const float scale = 0.08838834764831845f;  // 1/sqrt(128)

  for (int kb = 0; kb <= qb; ++kb) {
    // hint next K/V block into cache (global_prefetch_b8)
    if (kb + 1 <= qb) {
      __builtin_prefetch(Kp + (size_t)(kb + 1) * 64 * HD, 0, 3);
      __builtin_prefetch(V + (size_t)(kb + 1) * 64, 0, 3);
    }
    const int ntMax = (kb == qb) ? (wave + 1) : 4;
    v8f s[4];

    // S = Q K^T over this 64-key block (4 sub-tiles of 16 keys)
#pragma unroll
    for (int nt = 0; nt < 4; ++nt) {
      if (nt < ntMax) {
        const int key0 = kb * 64 + nt * 16;
        v8f sv = zero8();
#pragma unroll
        for (int c = 0; c < 4; ++c) {
          v16bf kf = load_fragB(Kp, key0, HD, c * 32);
          sv = wmma_bf16(qf[c], kf, sv);
        }
#pragma unroll
        for (int r = 0; r < 8; ++r) {
          float x = sv[r] * scale;
          if (kb == qb) {
            const int key = key0 + col;
            const int qr  = qRow0 + r + halfOff;
            if (key > qr) x = -__builtin_inff();
          }
          sv[r] = x;
        }
        s[nt] = sv;
      }
    }

    // online softmax: row max / exp / row sum via xor-shuffles within 16 lanes
    float fac[8];
#pragma unroll
    for (int r = 0; r < 8; ++r) {
      float tmax = -__builtin_inff();
#pragma unroll
      for (int nt = 0; nt < 4; ++nt)
        if (nt < ntMax) tmax = fmaxf(tmax, s[nt][r]);
#pragma unroll
      for (int off = 1; off <= 8; off <<= 1) tmax = fmaxf(tmax, __shfl_xor(tmax, off, 32));
      const float newm = fmaxf(mrow[r], tmax);
      const float f = __expf(mrow[r] - newm);
      float rsum = 0.f;
#pragma unroll
      for (int nt = 0; nt < 4; ++nt)
        if (nt < ntMax) {
          const float p = __expf(s[nt][r] - newm);
          s[nt][r] = p;
          rsum += p;
        }
#pragma unroll
      for (int off = 1; off <= 8; off <<= 1) rsum += __shfl_xor(rsum, off, 32);
      lrow[r] = lrow[r] * f + rsum;
      mrow[r] = newm;
      fac[r]  = f;
    }

    // rescale O accumulators
#pragma unroll
    for (int a = 0; a < 8; ++a)
#pragma unroll
      for (int r = 0; r < 8; ++r) acc[a][r] *= fac[r];

    // P (C-layout fp32) -> LDS bf16 row-major [16][64]; masked tiles -> 0
#pragma unroll
    for (int nt = 0; nt < 4; ++nt)
#pragma unroll
      for (int r = 0; r < 8; ++r) {
        const float p = (nt < ntMax) ? s[nt][r] : 0.f;
        pw[(r + halfOff) * 64 + nt * 16 + col] = f2bf(p);
      }
    asm volatile("s_wait_dscnt 0" ::: "memory");

    // O += P @ V   (V^T rows give contiguous B-fragments)
#pragma unroll
    for (int c = 0; c < 2; ++c) {
      v16bf pf = load_fragA(pw, 0, 64, c * 32);
#pragma unroll
      for (int a = 0; a < 8; ++a) {
        v16bf vf = load_fragB(V, a * 16, SEQ, kb * 64 + c * 32);
        acc[a] = wmma_bf16(pf, vf, acc[a]);
      }
    }
  }

  float inv[8];
#pragma unroll
  for (int r = 0; r < 8; ++r) inv[r] = 1.0f / lrow[r];
#pragma unroll
  for (int a = 0; a < 8; ++a)
#pragma unroll
    for (int r = 0; r < 8; ++r) {
      const int row = qRow0 + r + halfOff;
      const int cc  = h * HD + a * 16 + col;
      outb[(size_t)row * HID + cc] = f2bf(acc[a][r] * inv[r]);
    }
}

// ---------------------------------------------------------------------------
extern "C" void kernel_launch(void* const* d_in, const int* in_sizes, int n_in,
                              void* d_out, int out_size, void* d_ws, size_t ws_size,
                              hipStream_t stream) {
  (void)in_sizes; (void)n_in; (void)out_size; (void)ws_size;
  const float* x     = (const float*)d_in[0];
  const float* w_qkv = (const float*)d_in[1];
  const float* w_out = (const float*)d_in[2];
  // d_in[3] = causal mask: implied by kernel, unused
  float* out = (float*)d_out;

  char* ws = (char*)d_ws;
  const size_t SZ_XB   = (size_t)SEQ * HID * 2;      //  8 MiB
  const size_t SZ_WQKV = (size_t)THREE_H * HID * 2;  // 24 MiB
  const size_t SZ_WOUT = (size_t)HID * HID * 2;      //  8 MiB
  const size_t SZ_QKV  = (size_t)SEQ * THREE_H * 4;  // 48 MiB
  const size_t SZ_H    = (size_t)NH * SEQ * HD * 2;  //  8 MiB each

  size_t off = 0;
  unsigned short* xb    = (unsigned short*)(ws + off); off += SZ_XB;
  unsigned short* wqkvb = (unsigned short*)(ws + off); off += SZ_WQKV;
  unsigned short* woutb = (unsigned short*)(ws + off); off += SZ_WOUT;
  float*          qkv   = (float*)(ws + off);          off += SZ_QKV;
  unsigned short* qhb   = (unsigned short*)(ws + off); off += SZ_H;
  unsigned short* khb   = (unsigned short*)(ws + off); off += SZ_H;
  unsigned short* vtb   = (unsigned short*)(ws + off); off += SZ_H;
  unsigned short* attnb = (unsigned short*)(ws + off); off += SZ_XB;

  // 1) fp32 -> bf16
  f32_to_bf16_vec4<<<dim3((SEQ * HID / 4 + 255) / 256), 256, 0, stream>>>(x, xb, SEQ * HID / 4);
  f32_to_bf16_vec4<<<dim3((THREE_H * HID / 4 + 255) / 256), 256, 0, stream>>>(w_qkv, wqkvb, THREE_H * HID / 4);
  f32_to_bf16_vec4<<<dim3((HID * HID / 4 + 255) / 256), 256, 0, stream>>>(w_out, woutb, HID * HID / 4);

  // 2) qkv = x @ w_qkv^T  (fp32 accumulate, TDM-fed WMMA)
  gemm_bf16_nt<<<dim3(THREE_H / 128, SEQ / 128), 256, 0, stream>>>(xb, wqkvb, qkv, SEQ, THREE_H, HID);

  // 3) RMSNorm + RoPE -> per-head bf16 Q, K, V^T
  rmsnorm_rope<<<dim3(SEQ, NH), 128, 0, stream>>>(qkv, qhb, khb, vtb);

  // 4) causal flash attention -> attn output bf16 [SEQ][HID]
  flash_attn<<<dim3(SEQ / 64, NH), 128, 0, stream>>>(qhb, khb, vtb, attnb);

  // 5) out = attn @ w_out^T (fp32 straight to d_out)
  gemm_bf16_nt<<<dim3(HID / 128, SEQ / 128), 256, 0, stream>>>(attnb, woutb, out, SEQ, HID, HID);
}
